// MultiheadAttention_10213432230461
// MI455X (gfx1250) — matrix operand order repaired and checked by reading, not compiled, not run on hardware
//
#include <hip/hip_runtime.h>

// ---------------- types ----------------
typedef unsigned short u16;
typedef __attribute__((ext_vector_type(16))) __bf16       v16bf;
typedef __attribute__((ext_vector_type(8)))  float        v8f;
typedef __attribute__((ext_vector_type(4)))  unsigned int u32x4;

union VB { v16bf v; u32x4 q[2]; };

#define MODEL_DIM 1024
#define HEADS 16
#define HD 64
#define BATCH 2
#define SEQ 2048
#define MROWS (BATCH * SEQ)       // 4096
#define NQKV (3 * MODEL_DIM)      // 3072
#define ATTN_SCALE 0.125f         // 1/sqrt(64)

// ---------------- helpers ----------------
static __device__ inline u16 f2bf(float f) {
  union { float f; unsigned u; } x; x.f = f;
  unsigned r = x.u + 0x7fffu + ((x.u >> 16) & 1u);
  return (u16)(r >> 16);
}

// B-operand fragment: 8 bf16 at p, 8 bf16 at p+16 (per-lane)
static __device__ inline v16bf load16(const u16* p) {
  VB u;
  u.q[0] = *(const u32x4*)(p);
  u.q[1] = *(const u32x4*)(p + 16);
  return u.v;
}

// A-operand fragment: 8 bf16 at p, 8 bf16 at p+16 (per-lane)
static __device__ inline v16bf load_a(const u16* p) {
  VB u;
  u.q[0] = *(const u32x4*)(p);
  u.q[1] = *(const u32x4*)(p + 16);
  return u.v;
}

static __device__ inline v8f wmma_bf16(v16bf a, v16bf b, v8f c) {
  return __builtin_amdgcn_wmma_f32_16x16x32_bf16(
      /*neg_a=*/false, a, /*neg_b=*/false, b,
      /*c_mod=*/(short)0, c, /*reuse_a=*/false, /*reuse_b=*/false);
}

static __device__ inline v8f vzero8() {
  v8f z;
#pragma unroll
  for (int i = 0; i < 8; i++) z[i] = 0.0f;
  return z;
}

// CDNA5 async global->LDS copy (ASYNCcnt-tracked), GV addressing mode.
static __device__ inline void async_cp_b128(u16* lds_p, const u16* g) {
  unsigned l = (unsigned)(size_t)lds_p;    // low 32 bits = LDS byte offset
  asm volatile("global_load_async_to_lds_b128 %0, %1, off"
               :: "v"(l), "v"(g) : "memory");
}
static __device__ inline void wait_async0() {
  asm volatile("s_wait_asynccnt 0" ::: "memory");
}

// ---------------- prep kernels ----------------
__global__ void mha_cvt_bf16(const float* __restrict__ src, u16* __restrict__ dst, int n) {
  int i = blockIdx.x * blockDim.x + threadIdx.x;
  if (i < n) dst[i] = f2bf(src[i]);
}

// src is R x C (row-major). dst is C x R (K-major weights): dst[c*R + r] = src[r*C + c]
__global__ void mha_transpose_bf16(const float* __restrict__ src, u16* __restrict__ dst, int R, int C) {
  int i = blockIdx.x * blockDim.x + threadIdx.x;
  if (i >= R * C) return;
  int r = i % R;
  int c = i / R;
  dst[i] = f2bf(src[r * C + c]);
}

// ---------------- shared GEMM core ----------------
// 8-wave workgroup, WG tile 128x64. B panel (64 cols x 32 k, K-major) staged in a
// double-buffered 2x4KB LDS tile via async copies; each wave computes its own 16x64
// slice with WMMA. A fragments double-buffered in registers.
static __device__ inline void gemm_tile_core(
    const u16* __restrict__ arow,   // per-lane A row pointer (already + half*8)
    const u16* __restrict__ wT,     // K-major weights [N][K]
    int nb, u16* Bt /*shared: 2*64*32 u16*/, int lane, int tid, v8f acc[4]) {
  const int half = lane >> 4;
  const int ln   = lane & 15;

  // staging map: 256 threads x 16B covers 64x32 bf16 tile
  const int sn = tid >> 2;          // 0..63 : column of wT
  const int sp = tid & 3;           // 0..3  : 8-element chunk of k
  const u16* gstage = wT + (nb + sn) * MODEL_DIM + sp * 8;
  u16* lstage = Bt + sn * 32 + sp * 8;

  async_cp_b128(lstage, gstage);            // prologue: k=0 -> buf0
  v16bf a_next = load_a(arow);              // k=0 A fragment

  for (int k = 0; k < MODEL_DIM; k += 32) {
    const int cur = (k >> 5) & 1;
    wait_async0();
    __syncthreads();                        // buf[cur] fully staged; buf[cur^1] free
    if (k + 32 < MODEL_DIM)
      async_cp_b128(lstage + (cur ^ 1) * 2048, gstage + (k + 32));

    v16bf a = a_next;
    if (k + 32 < MODEL_DIM) a_next = load_a(arow + k + 32);

    const u16* lb = Bt + cur * 2048;
#pragma unroll
    for (int t = 0; t < 4; t++) {
      const u16* bp = lb + (t * 16 + ln) * 32 + half * 8;
      acc[t] = wmma_bf16(a, load16(bp), acc[t]);
    }
  }
}

// ---------------- QKV projection GEMM ----------------
__global__ __launch_bounds__(256) void mha_qkv_gemm(
    const u16* __restrict__ xb, const u16* __restrict__ wT, const float* __restrict__ bias,
    u16* __restrict__ Qb, u16* __restrict__ Kb, u16* __restrict__ VT) {
  __shared__ u16 Bt[2 * 64 * 32];
  const int tid  = threadIdx.x;
  const int lane = tid & 31;
  const int wave = tid >> 5;
  const int half = lane >> 4;
  const int ln   = lane & 15;
  const int mbase = blockIdx.x * 128 + wave * 16;
  const int nb    = blockIdx.y * 64;

  const u16* arow = xb + (mbase + ln) * MODEL_DIM + half * 8;

  v8f acc[4];
#pragma unroll
  for (int t = 0; t < 4; t++) acc[t] = vzero8();

  gemm_tile_core(arow, wT, nb, Bt, lane, tid, acc);

#pragma unroll
  for (int t = 0; t < 4; t++) {
    const int n = nb + t * 16 + ln;
    const int which = n >> 10;        // 0=Q 1=K 2=V
    const int rem   = n & 1023;
    const int head  = rem >> 6;
    const int col   = rem & 63;
    const float bv  = bias[n];
#pragma unroll
    for (int r = 0; r < 8; r++) {
      const int row  = r + half * 8;
      const int mrow = mbase + row;
      const int b    = mrow >> 11;     // /2048
      const int s    = mrow & 2047;
      const int bh   = b * HEADS + head;
      const u16 hv   = f2bf(acc[t][r] + bv);
      if (which == 0)      Qb[(bh * SEQ + s) * HD + col] = hv;
      else if (which == 1) Kb[(bh * SEQ + s) * HD + col] = hv;
      else                 VT[(bh * HD + col) * SEQ + s] = hv;
    }
  }
}

// ---------------- flash attention ----------------
// One wave per (batch*head, 16-query tile). Online softmax over 64 blocks of 32 keys.
__global__ __launch_bounds__(32) void mha_flash_attn(
    const u16* __restrict__ Qb, const u16* __restrict__ Kb, const u16* __restrict__ VT,
    u16* __restrict__ zb) {
  __shared__ u16 Pl[16 * 32];

  const int lane = threadIdx.x;
  const int half = lane >> 4;
  const int ln   = lane & 15;
  const int qt   = blockIdx.x;              // query tile
  const int bh   = blockIdx.y;              // b*HEADS + head
  const int b    = bh >> 4;
  const int head = bh & 15;

  const u16* Q = Qb + (bh * SEQ + qt * 16) * HD;
  const u16* K = Kb + bh * SEQ * HD;
  const u16* V = VT + bh * HD * SEQ;

  // Q A-operand fragments for hd 0..31 and 32..63 (loaded once)
  const v16bf a0 = load_a(Q + ln * HD + half * 8);
  const v16bf a1 = load_a(Q + ln * HD + 32 + half * 8);

  v8f z[4];
#pragma unroll
  for (int t = 0; t < 4; t++) z[t] = vzero8();
  float m[8], L[8];
#pragma unroll
  for (int r = 0; r < 8; r++) { m[r] = -1e30f; L[r] = 0.0f; }

  for (int kb = 0; kb < SEQ; kb += 32) {
    // ---- scores: S[16q x 32keys] = Q @ K^T (two 16-key groups) ----
    v8f s0 = vzero8(), s1 = vzero8();
    {
      const u16* k0 = K + (kb + ln) * HD + half * 8;       // key group 0
      const u16* k1 = K + (kb + 16 + ln) * HD + half * 8;  // key group 1
      s0 = wmma_bf16(a0, load16(k0),       s0);
      s0 = wmma_bf16(a1, load16(k0 + 32),  s0);
      s1 = wmma_bf16(a0, load16(k1),       s1);
      s1 = wmma_bf16(a1, load16(k1 + 32),  s1);
    }

    // V fragments loaded early so they overlap the softmax VALU/TRANS chain
    v16bf vf[4];
#pragma unroll
    for (int t = 0; t < 4; t++)
      vf[t] = load16(V + (t * 16 + ln) * SEQ + kb + half * 8);

    // ---- online softmax (per-row segmented reductions over 16-lane halves) ----
    float scf[8];
#pragma unroll
    for (int r = 0; r < 8; r++) {
      float v0 = s0[r] * ATTN_SCALE;
      float v1 = s1[r] * ATTN_SCALE;
      float mx = fmaxf(v0, v1);
      mx = fmaxf(mx, __shfl_xor(mx, 8, 16));
      mx = fmaxf(mx, __shfl_xor(mx, 4, 16));
      mx = fmaxf(mx, __shfl_xor(mx, 2, 16));
      mx = fmaxf(mx, __shfl_xor(mx, 1, 16));
      const float mn = fmaxf(m[r], mx);
      const float sc = __expf(m[r] - mn);
      const float p0 = __expf(v0 - mn);
      const float p1 = __expf(v1 - mn);
      float rs = p0 + p1;
      rs += __shfl_xor(rs, 8, 16);
      rs += __shfl_xor(rs, 4, 16);
      rs += __shfl_xor(rs, 2, 16);
      rs += __shfl_xor(rs, 1, 16);
      L[r] = L[r] * sc + rs;
      m[r] = mn;
      scf[r] = sc;
      const int row = r + half * 8;
      Pl[row * 32 + ln]      = f2bf(p0);
      Pl[row * 32 + 16 + ln] = f2bf(p1);
    }
    __syncthreads();   // single-wave WG: orders DS stores before DS loads

    // rescale running Z accumulator
#pragma unroll
    for (int t = 0; t < 4; t++)
#pragma unroll
      for (int r = 0; r < 8; r++) z[t][r] *= scf[r];

    // ---- Z += P @ V : A = P(16x32) from LDS, B = V^T tiles (32keys x 16cols) ----
    const v16bf pa = load_a((const u16*)Pl + ln * 32 + half * 8);
#pragma unroll
    for (int t = 0; t < 4; t++) z[t] = wmma_bf16(pa, vf[t], z[t]);
    __syncthreads();   // P tile consumed before next iteration overwrites it
  }

  // ---- finalize: divide by row sums, store z as bf16 into [B,S,H*64] ----
#pragma unroll
  for (int r = 0; r < 8; r++) {
    const float inv = (L[r] > 0.0f) ? (1.0f / L[r]) : 0.0f;
    const int row = r + half * 8;
    const int s   = qt * 16 + row;
    u16* zr = zb + (b * SEQ + s) * MODEL_DIM + head * HD;
#pragma unroll
    for (int t = 0; t < 4; t++) zr[t * 16 + ln] = f2bf(z[t][r] * inv);
  }
}

// ---------------- output projection GEMM ----------------
__global__ __launch_bounds__(256) void mha_out_gemm(
    const u16* __restrict__ zbb, const u16* __restrict__ wT, const float* __restrict__ bias,
    float* __restrict__ out) {
  __shared__ u16 Bt[2 * 64 * 32];
  const int tid  = threadIdx.x;
  const int lane = tid & 31;
  const int wave = tid >> 5;
  const int half = lane >> 4;
  const int ln   = lane & 15;
  const int mbase = blockIdx.x * 128 + wave * 16;
  const int nb    = blockIdx.y * 64;

  const u16* arow = zbb + (mbase + ln) * MODEL_DIM + half * 8;

  v8f acc[4];
#pragma unroll
  for (int t = 0; t < 4; t++) acc[t] = vzero8();

  gemm_tile_core(arow, wT, nb, Bt, lane, tid, acc);

#pragma unroll
  for (int t = 0; t < 4; t++) {
    const int n = nb + t * 16 + ln;
    const float bv = bias[n];
#pragma unroll
    for (int r = 0; r < 8; r++) {
      const int row = r + half * 8;
      out[(mbase + row) * MODEL_DIM + n] = acc[t][r] + bv;
    }
  }
}

// ---------------- launch ----------------
extern "C" void kernel_launch(void* const* d_in, const int* in_sizes, int n_in,
                              void* d_out, int out_size, void* d_ws, size_t ws_size,
                              hipStream_t stream) {
  (void)in_sizes; (void)n_in; (void)out_size; (void)ws_size;
  const float* x     = (const float*)d_in[0];   // [2,2048,1024]
  const float* w_qkv = (const float*)d_in[1];   // [1024,3072]
  const float* b_qkv = (const float*)d_in[2];   // [3072]
  const float* w_out = (const float*)d_in[3];   // [1024,1024]
  const float* b_out = (const float*)d_in[4];   // [1024]
  float* out = (float*)d_out;                   // [2,2048,1024]

  // workspace carve (all bf16), 48 MB total
  u16* xb  = (u16*)d_ws;                                  // 4096*1024
  u16* wqT = xb  + (size_t)MROWS * MODEL_DIM;             // 3072*1024 (K-major)
  u16* woT = wqT + (size_t)NQKV * MODEL_DIM;              // 1024*1024 (K-major)
  u16* Qb  = woT + (size_t)MODEL_DIM * MODEL_DIM;         // [B,H,S,64]
  u16* Kb  = Qb  + (size_t)BATCH * HEADS * SEQ * HD;
  u16* VT  = Kb  + (size_t)BATCH * HEADS * SEQ * HD;      // [B,H,64,S]
  u16* zb  = VT  + (size_t)BATCH * HEADS * SEQ * HD;      // [B,S,1024]

  // 1) precision/layout prep
  {
    int n = MROWS * MODEL_DIM;
    mha_cvt_bf16<<<(n + 255) / 256, 256, 0, stream>>>(x, xb, n);
  }
  {
    int n = MODEL_DIM * NQKV;
    mha_transpose_bf16<<<(n + 255) / 256, 256, 0, stream>>>(w_qkv, wqT, MODEL_DIM, NQKV);
  }
  {
    int n = MODEL_DIM * MODEL_DIM;
    mha_transpose_bf16<<<(n + 255) / 256, 256, 0, stream>>>(w_out, woT, MODEL_DIM, MODEL_DIM);
  }

  // 2) QKV projection: WG tile 128x64, 8 waves sharing async-staged B panel
  mha_qkv_gemm<<<dim3(MROWS / 128, NQKV / 64), 256, 0, stream>>>(xb, wqT, b_qkv, Qb, Kb, VT);

  // 3) flash attention (128 q-tiles x 32 batch-heads)
  mha_flash_attn<<<dim3(SEQ / 16, BATCH * HEADS), 32, 0, stream>>>(Qb, Kb, VT, zb);

  // 4) output projection
  mha_out_gemm<<<dim3(MROWS / 128, MODEL_DIM / 64), 256, 0, stream>>>(zb, woT, b_out, out);
}
